// SetAbstraction_15479062135522
// MI455X (gfx1250) — matrix operand order repaired and checked by reading, not compiled, not run on hardware
//
#include <hip/hip_runtime.h>

#define BSZ    8
#define NPTS   4096
#define NCENT  1024
#define NSAMP  32
#define DFEAT  64

typedef _Float16 half16 __attribute__((ext_vector_type(16)));
typedef float    float8 __attribute__((ext_vector_type(8)));

union F16x16 { half16 h; unsigned int u[8]; };
union F32x8  { float8 f; float e[8]; };

// ---------------------------------------------------------------------------
// utility: wave32 max-reduce of a packed u64 key
// ---------------------------------------------------------------------------
__device__ __forceinline__ unsigned long long wave_max64(unsigned long long k) {
#pragma unroll
  for (int off = 16; off > 0; off >>= 1) {
    unsigned long long o = __shfl_xor(k, off, 32);
    if (o > k) k = o;
  }
  return k;
}

// ---------------------------------------------------------------------------
// Kernel 1: farthest point sampling. 1 workgroup (1024 thr = 32 waves) / batch.
// xyz cached in LDS; per-thread masks & coords in registers.
// argmax key = (f32bits(d) << 32) | (NPTS-1-p)  -> first-index tie-break.
// Writes cent_xyz straight into d_out[0 .. B*NCENT*3).
// ---------------------------------------------------------------------------
__global__ __launch_bounds__(1024) void fps_kernel(
    const float* __restrict__ xyz, const int* __restrict__ init_idx,
    float* __restrict__ out) {
  __shared__ float sx[NPTS], sy[NPTS], sz[NPTS];
  __shared__ unsigned long long wred[32];
  __shared__ int s_choice;

  const int b    = blockIdx.x;
  const int tid  = threadIdx.x;
  const int lane = tid & 31, wid = tid >> 5;
  const float* xb = xyz + (size_t)b * NPTS * 3;

  float lx[4], ly[4], lz[4], msk[4];
#pragma unroll
  for (int j = 0; j < 4; ++j) {
    int p = tid + j * 1024;
    float x = xb[p * 3 + 0], y = xb[p * 3 + 1], z = xb[p * 3 + 2];
    lx[j] = x; ly[j] = y; lz[j] = z; msk[j] = 1.0f;
    sx[p] = x; sy[p] = y; sz[p] = z;
  }
  __syncthreads();

  int cur = init_idx[b];
  float cx = sx[cur], cy = sy[cur], cz = sz[cur];
  if (tid == 0) {
    out[(b * NCENT) * 3 + 0] = cx;
    out[(b * NCENT) * 3 + 1] = cy;
    out[(b * NCENT) * 3 + 2] = cz;
  }

  for (int t = 1; t < NCENT; ++t) {
    unsigned long long best = 0ull;
#pragma unroll
    for (int j = 0; j < 4; ++j) {
      int p = tid + j * 1024;
      float dx = lx[j] - cx, dy = ly[j] - cy, dz = lz[j] - cz;
      float d = sqrtf(dx * dx + dy * dy + dz * dz) * msk[j];
      msk[j] = fminf(d * msk[j] * 1e11f, msk[j]);   // reference mask quirk
      unsigned long long key =
          ((unsigned long long)__float_as_uint(d) << 32) |
          (unsigned int)(NPTS - 1 - p);
      if (key > best) best = key;
    }
    best = wave_max64(best);
    if (lane == 0) wred[wid] = best;
    __syncthreads();
    if (wid == 0) {
      unsigned long long k = wave_max64(wred[lane]);
      if (lane == 0) {
        int idx = NPTS - 1 - (int)(unsigned int)(k & 0xffffffffu);
        s_choice = idx;
        out[(b * NCENT + t) * 3 + 0] = sx[idx];
        out[(b * NCENT + t) * 3 + 1] = sy[idx];
        out[(b * NCENT + t) * 3 + 2] = sz[idx];
      }
    }
    __syncthreads();
    cur = s_choice;
    cx = sx[cur]; cy = sy[cur]; cz = sz[cur];
  }
}

// ---------------------------------------------------------------------------
// Kernel 2: ball query (stable top-32 of clipped distance). 1 wave / centroid,
// 2 waves / block. Distances as u32 float-bits in LDS; 32 min-extraction
// rounds with (d, idx) tie-break == stable argsort[:32].
// ---------------------------------------------------------------------------
__global__ __launch_bounds__(64) void ballq_kernel(
    const float* __restrict__ xyz, const float* __restrict__ cent_xyz,
    int* __restrict__ group_idx) {
  __shared__ unsigned int sd[2][NPTS];
  const int lane = threadIdx.x & 31, wid = threadIdx.x >> 5;
  const int cid  = blockIdx.x * 2 + wid;
  const int b    = cid >> 10;
  const float cx = cent_xyz[cid * 3 + 0];
  const float cy = cent_xyz[cid * 3 + 1];
  const float cz = cent_xyz[cid * 3 + 2];
  const float* xb = xyz + (size_t)b * NPTS * 3;
  const float clip = 0.2f * 0.2f;            // radius**2 quirk (vs raw dist)

#pragma unroll 4
  for (int j = 0; j < 128; ++j) {
    int p = lane + j * 32;
    float dx = xb[p * 3 + 0] - cx, dy = xb[p * 3 + 1] - cy, dz = xb[p * 3 + 2] - cz;
    float d = fminf(sqrtf(dx * dx + dy * dy + dz * dz), clip);
    sd[wid][p] = __float_as_uint(d);
  }
  for (int r = 0; r < NSAMP; ++r) {
    unsigned int du = 0xffffffffu; int pi = 0;
    for (int j = 0; j < 128; ++j) {          // lane owns p = lane + 32*j
      int p = lane + j * 32;
      unsigned int v = sd[wid][p];
      if (v < du) { du = v; pi = p; }
    }
#pragma unroll
    for (int off = 16; off > 0; off >>= 1) {
      unsigned int od = __shfl_xor(du, off, 32);
      int opi         = __shfl_xor(pi, off, 32);
      if (od < du || (od == du && opi < pi)) { du = od; pi = opi; }
    }
    if ((pi & 31) == lane) sd[wid][pi] = 0x7f800000u;  // mark chosen (+inf)
    if (lane == 0) group_idx[cid * NSAMP + r] = pi;
  }
}

// ---------------------------------------------------------------------------
// Kernel 3: gather + 3-layer MLP via v_wmma_f32_16x16x32_f16 + max-pool.
// 1 wave = 1 centroid (32 rows = 2 M-tiles). Activations in LDS (f16,
// row stride 130 halves), weights in LDS (f16, [n][k] stride 100 halves).
// ---------------------------------------------------------------------------
#define ACT_STRIDE 130   // halves; 65 dwords -> conflict-free row access
#define W_STRIDE   100   // halves per output channel

template <int K, int RK, int N, bool LAST>
__device__ __forceinline__ void mlp_layer(
    const float* __restrict__ W, const float* __restrict__ bias,
    const float* __restrict__ gamma, const float* __restrict__ beta,
    const float* __restrict__ mean, const float* __restrict__ var,
    _Float16* __restrict__ Wb, float* __restrict__ s_scale,
    float* __restrict__ s_shift, _Float16* __restrict__ aw,
    float* __restrict__ outp) {
  const int tid = threadIdx.x;
  const int lane = tid & 31;
  __syncthreads();  // previous layer's Wb readers done
  for (int idx = tid; idx < N * K; idx += 128) {
    int n = idx / K, k = idx - n * K;
    float w = (k < RK) ? W[k * N + n] : 0.f;   // zero-pad K to mult. of 32
    Wb[n * W_STRIDE + k] = (_Float16)w;
  }
  for (int n = tid; n < N; n += 128) {
    float sc = gamma[n] * rsqrtf(var[n] + 1e-3f);
    s_scale[n] = sc;
    s_shift[n] = (bias[n] - mean[n]) * sc + beta[n];
  }
  __syncthreads();

  constexpr int KT = K / 32, NT = N / 16;
  const int r15 = lane & 15, hi = lane >> 4;
  const unsigned int* actw = (const unsigned int*)aw;
  const unsigned int* Wbw  = (const unsigned int*)Wb;

  // Preload all A fragments (ISA 16-bit A 16x32 layout) so we can overwrite
  // the activation buffer with this layer's outputs.
  F16x16 afr[2][KT];
#pragma unroll
  for (int mt = 0; mt < 2; ++mt) {
    int row = 16 * mt + r15;
#pragma unroll
    for (int kt = 0; kt < KT; ++kt)
#pragma unroll
      for (int v = 0; v < 8; ++v)
        afr[mt][kt].u[v] =
            actw[row * (ACT_STRIDE / 2) + 16 * kt + ((v < 4) ? 0 : 8) + hi * 4 + (v & 3)];
  }

#pragma unroll
  for (int nt = 0; nt < NT; ++nt) {
    const int ncol = 16 * nt + r15;
    const float sc = s_scale[ncol], sh = s_shift[ncol];
    float pmax = 0.f;   // ReLU outputs are >= 0
#pragma unroll
    for (int mt = 0; mt < 2; ++mt) {
      F32x8 c;
#pragma unroll
      for (int v = 0; v < 8; ++v) c.e[v] = 0.f;
#pragma unroll
      for (int kt = 0; kt < KT; ++kt) {
        F16x16 bfr;  // B 32x16 f16: lane=col, dword v = K pair (hi half +8, v>=4 +16)
#pragma unroll
        for (int v = 0; v < 8; ++v)
          bfr.u[v] = Wbw[ncol * (W_STRIDE / 2) + 16 * kt + ((v < 4) ? 0 : 8) + hi * 4 + (v & 3)];
        c.f = __builtin_amdgcn_wmma_f32_16x16x32_f16(
            false, afr[mt][kt].h, false, bfr.h, (short)0, c.f, false, false);
      }
      if (!LAST) {
#pragma unroll
        for (int v = 0; v < 8; ++v) {
          float y = fmaxf(c.e[v] * sc + sh, 0.f);      // folded BN + ReLU
          int row = 16 * mt + v + 8 * hi;              // C layout: M = v+8*hi
          aw[row * ACT_STRIDE + ncol] = (_Float16)y;
        }
      } else {
#pragma unroll
        for (int v = 0; v < 8; ++v)
          pmax = fmaxf(pmax, fmaxf(c.e[v] * sc + sh, 0.f));
      }
    }
    if (LAST) {  // max over 32 samples: other 16 rows live in the other half-wave
      pmax = fmaxf(pmax, __shfl_xor(pmax, 16, 32));
      if (lane < 16) outp[16 * nt + r15] = pmax;
    }
  }
}

__global__ __launch_bounds__(128) void mlp_wmma_kernel(
    const float* __restrict__ xyz, const float* __restrict__ points,
    const int* __restrict__ group_idx,
    const float* __restrict__ W0, const float* __restrict__ b0,
    const float* __restrict__ g0, const float* __restrict__ be0,
    const float* __restrict__ m0, const float* __restrict__ v0,
    const float* __restrict__ W1, const float* __restrict__ b1,
    const float* __restrict__ g1, const float* __restrict__ be1,
    const float* __restrict__ m1, const float* __restrict__ v1,
    const float* __restrict__ W2, const float* __restrict__ b2,
    const float* __restrict__ g2, const float* __restrict__ be2,
    const float* __restrict__ m2, const float* __restrict__ v2,
    float* __restrict__ out) {
  __shared__ _Float16 Wb[128 * W_STRIDE];        // 25.0 KB
  __shared__ float s_scale[128], s_shift[128];   //  1.0 KB
  __shared__ _Float16 act[4][32 * ACT_STRIDE];   // 32.5 KB

  const int lane = threadIdx.x & 31, wid = threadIdx.x >> 5;
  const int cid  = blockIdx.x * 4 + wid;
  const int b    = cid >> 10;
  _Float16* aw = act[wid];

  // Gather: lane = sample row; x = [xyz(3) | feats(64) | zero-pad -> 96]
  {
    int g = group_idx[cid * NSAMP + lane];
    const float* px = xyz    + ((size_t)b * NPTS + g) * 3;
    const float* pf = points + ((size_t)b * NPTS + g) * DFEAT;
    _Float16* row = aw + lane * ACT_STRIDE;
    row[0] = (_Float16)px[0];
    row[1] = (_Float16)px[1];
    row[2] = (_Float16)px[2];
#pragma unroll
    for (int f = 0; f < DFEAT; f += 4) {
      float4 q = *(const float4*)(pf + f);
      row[3 + f + 0] = (_Float16)q.x;
      row[3 + f + 1] = (_Float16)q.y;
      row[3 + f + 2] = (_Float16)q.z;
      row[3 + f + 3] = (_Float16)q.w;
    }
#pragma unroll
    for (int k = 67; k < 96; ++k) row[k] = (_Float16)0.f;
  }

  float* outp = out + (size_t)BSZ * NCENT * 3 + (size_t)cid * 128;
  mlp_layer<96, 67, 64, false>(W0, b0, g0, be0, m0, v0, Wb, s_scale, s_shift, aw, outp);
  mlp_layer<64, 64, 64, false>(W1, b1, g1, be1, m1, v1, Wb, s_scale, s_shift, aw, outp);
  mlp_layer<64, 64, 128, true>(W2, b2, g2, be2, m2, v2, Wb, s_scale, s_shift, aw, outp);
}

// ---------------------------------------------------------------------------
extern "C" void kernel_launch(void* const* d_in, const int* in_sizes, int n_in,
                              void* d_out, int out_size, void* d_ws, size_t ws_size,
                              hipStream_t stream) {
  (void)in_sizes; (void)n_in; (void)out_size; (void)ws_size;
  const float* xyz      = (const float*)d_in[0];
  const float* points   = (const float*)d_in[1];
  const int*   init_idx = (const int*)d_in[2];
  const float* W0 = (const float*)d_in[3],  *b0 = (const float*)d_in[4];
  const float* g0 = (const float*)d_in[5],  *be0 = (const float*)d_in[6];
  const float* m0 = (const float*)d_in[7],  *v0 = (const float*)d_in[8];
  const float* W1 = (const float*)d_in[9],  *b1 = (const float*)d_in[10];
  const float* g1 = (const float*)d_in[11], *be1 = (const float*)d_in[12];
  const float* m1 = (const float*)d_in[13], *v1 = (const float*)d_in[14];
  const float* W2 = (const float*)d_in[15], *b2 = (const float*)d_in[16];
  const float* g2 = (const float*)d_in[17], *be2 = (const float*)d_in[18];
  const float* m2 = (const float*)d_in[19], *v2 = (const float*)d_in[20];

  float* out      = (float*)d_out;
  int*   grp_idx  = (int*)d_ws;                    // [B*NCENT*NSAMP]

  fps_kernel<<<BSZ, 1024, 0, stream>>>(xyz, init_idx, out);
  ballq_kernel<<<(BSZ * NCENT) / 2, 64, 0, stream>>>(xyz, out, grp_idx);
  mlp_wmma_kernel<<<(BSZ * NCENT) / 4, 128, 0, stream>>>(
      xyz, points, grp_idx,
      W0, b0, g0, be0, m0, v0,
      W1, b1, g1, be1, m1, v1,
      W2, b2, g2, be2, m2, v2,
      out);
}
